// RotaryMultiHeadAttention_83227876262568
// MI455X (gfx1250) — matrix operand order repaired and checked by reading, not compiled, not run on hardware
//
#include <hip/hip_runtime.h>
#include <hip/hip_bf16.h>

// ---------------------------------------------------------------------------
// Rotary multi-head attention on gfx1250 (MI455X), f16 WMMA / f32 accumulate.
// Flash attention stages K/V blocks into LDS with async global->LDS copies
// (ASYNCcnt) shared by 4 waves per workgroup, double-buffered.
// ---------------------------------------------------------------------------

typedef __attribute__((ext_vector_type(16))) _Float16 v16h;
typedef __attribute__((ext_vector_type(4)))  _Float16 v4h;
typedef __attribute__((ext_vector_type(8)))  float    v8f;
typedef __attribute__((ext_vector_type(4)))  int      i32x4;

#define GLOBAL_AS __attribute__((address_space(1)))
#define LDS_AS    __attribute__((address_space(3)))

union Frag {                // one WMMA 16-bit operand: 8 VGPRs / lane
    v16h  h;
    uint4 u[2];
};

#define D_MODEL 1024
#define NHEADS  16
#define HDIM    64
#define SEQ     2048
#define BATCH   2
#define MTOT    (BATCH * SEQ)      // 4096 tokens

#if defined(__has_builtin)
#if __has_builtin(__builtin_amdgcn_global_load_async_to_lds_b128)
#define HAVE_ASYNC_LDS 1
#endif
#endif
#ifndef HAVE_ASYNC_LDS
#define HAVE_ASYNC_LDS 0
#endif

__device__ __forceinline__ v8f wmma16(const Frag& a, const Frag& b, v8f c) {
    // v_wmma_f32_16x16x32_f16  (8-arg form, probe-confirmed)
    return __builtin_amdgcn_wmma_f32_16x16x32_f16(
        false, a.h, false, b.h, (short)0, c, false, false);
}

__device__ __forceinline__ float redmax16(float v) {
    v = fmaxf(v, __shfl_xor(v, 1, 16));
    v = fmaxf(v, __shfl_xor(v, 2, 16));
    v = fmaxf(v, __shfl_xor(v, 4, 16));
    v = fmaxf(v, __shfl_xor(v, 8, 16));
    return v;
}
__device__ __forceinline__ float redsum16(float v) {
    v += __shfl_xor(v, 1, 16);
    v += __shfl_xor(v, 2, 16);
    v += __shfl_xor(v, 4, 16);
    v += __shfl_xor(v, 8, 16);
    return v;
}

// 16-byte global -> LDS copy: async on CDNA5 (ASYNCcnt), sync fallback.
__device__ __forceinline__ void cp16_g2l(_Float16* l, const _Float16* g) {
#if HAVE_ASYNC_LDS
    __builtin_amdgcn_global_load_async_to_lds_b128(
        (GLOBAL_AS i32x4*)g, (LDS_AS i32x4*)l, 0, 0);
#else
    *(uint4*)l = *(const uint4*)g;
#endif
}

__device__ __forceinline__ void wait_async_lds() {
#if HAVE_ASYNC_LDS
#if defined(__has_builtin) && __has_builtin(__builtin_amdgcn_s_wait_asynccnt)
    __builtin_amdgcn_s_wait_asynccnt(0);
#else
    asm volatile("s_wait_asynccnt 0" ::: "memory");
#endif
#endif
}

// ---------------------------------------------------------------------------
// f32 -> f16 conversion (4 elems / thread, vector store)
// ---------------------------------------------------------------------------
__global__ __launch_bounds__(256)
void cvt_f32_f16(const float* __restrict__ in, _Float16* __restrict__ out, int n) {
    int i = (blockIdx.x * 256 + threadIdx.x) * 4;
    if (i + 3 < n) {
        float4 v = *(const float4*)(in + i);
        v4h h;
        h[0] = (_Float16)v.x; h[1] = (_Float16)v.y;
        h[2] = (_Float16)v.z; h[3] = (_Float16)v.w;
        *(v4h*)(out + i) = h;
    }
}

// ---------------------------------------------------------------------------
// Projection GEMM: out[m,n] = X[m,:] . W[n,:] + bias[n]   (out = X W^T + b)
// block = 128 threads = 4 waves; each wave -> 16(M) x 64(N) tile (one head).
// grid = (MTOT/16, D_MODEL/256)
// mode 0: f32 out, row-major [m][n]                    (final projection)
// mode 1: RoPE applied, f16 out [b,h,s,d]              (Q and K)
// mode 2: f16 out transposed [b,h,d,s]                 (V)
// ---------------------------------------------------------------------------
__global__ __launch_bounds__(128)
void proj_gemm(const _Float16* __restrict__ X,   // [MTOT][1024]
               const _Float16* __restrict__ W,   // [1024][1024] row-major
               const float*    __restrict__ bias,
               const int*      __restrict__ pos, // [MTOT] (mode 1), else null
               _Float16* __restrict__ out16,
               float*    __restrict__ out32,
               int mode) {
    const int wave = threadIdx.x >> 5;
    const int lane = threadIdx.x & 31;
    const int half = lane >> 4;
    const int ln   = lane & 15;

    const int mbase = blockIdx.x * 16;
    const int nbase = blockIdx.y * 256 + wave * 64;

    v8f c[4] = {};

    const _Float16* Xrow = X + (size_t)(mbase + ln) * D_MODEL;

#pragma unroll 4
    for (int ks = 0; ks < D_MODEL / 32; ++ks) {
        Frag a;
        const int k0 = ks * 32 + 8 * half;
        a.u[0] = *(const uint4*)(Xrow + k0);
        a.u[1] = *(const uint4*)(Xrow + k0 + 16);
#pragma unroll
        for (int t = 0; t < 4; ++t) {
            Frag b;
            const _Float16* Wp =
                W + (size_t)(nbase + t * 16 + ln) * D_MODEL + ks * 32 + 8 * half;
            b.u[0] = *(const uint4*)(Wp);
            b.u[1] = *(const uint4*)(Wp + 16);
            c[t] = wmma16(a, b, c[t]);
        }
    }

    // bias (column n is fixed per lane per tile)
#pragma unroll
    for (int t = 0; t < 4; ++t) {
        const float bv = bias[nbase + t * 16 + ln];
#pragma unroll
        for (int r = 0; r < 8; ++r) c[t][r] += bv;
    }

    if (mode == 0) {
#pragma unroll
        for (int t = 0; t < 4; ++t)
#pragma unroll
            for (int r = 0; r < 8; ++r) {
                const int m = mbase + r + 8 * half;
                out32[(size_t)m * D_MODEL + nbase + t * 16 + ln] = c[t][r];
            }
        return;
    }

    const int head = nbase >> 6;       // N tile is head-aligned

    if (mode == 2) {                    // V: store transposed [b,h,d,s]
#pragma unroll
        for (int t = 0; t < 4; ++t)
#pragma unroll
            for (int r = 0; r < 8; ++r) {
                const int m = mbase + r + 8 * half;
                const int b = m >> 11, s = m & (SEQ - 1);
                const int d = t * 16 + ln;
                out16[((size_t)(b * NHEADS + head) * HDIM + d) * SEQ + s] =
                    (_Float16)c[t][r];
            }
        return;
    }

    // mode 1: RoPE.  pair (d, d+32) lives in tiles (t, t+2) of the SAME lane.
    const float L2B_32 = 0.4152410119f;        // log2(10000)/32
    float posr[8];
#pragma unroll
    for (int r = 0; r < 8; ++r) posr[r] = (float)pos[mbase + r + 8 * half];

#pragma unroll
    for (int t = 0; t < 2; ++t) {
        const int fi = 8 * t + (ln >> 1);               // d/2, d = 16t+ln < 32
        const float invf_lo = exp2f(-(float)fi * L2B_32);
        const float invf_hi = exp2f(-(float)(fi + 16) * L2B_32);
#pragma unroll
        for (int r = 0; r < 8; ++r) {
            const float a1 = posr[r] * invf_lo;
            const float a2 = posr[r] * invf_hi;
            const float cs1 = __cosf(a1), sn1 = __sinf(a1);
            const float cs2 = __cosf(a2), sn2 = __sinf(a2);
            const float x1 = c[t][r], x2 = c[t + 2][r];
            const float o1 = x1 * cs1 - x2 * sn1;       // d < 32: rot = -x2
            const float o2 = x2 * cs2 + x1 * sn2;       // d >= 32: rot = +x1
            const int m = mbase + r + 8 * half;
            const int b = m >> 11, s = m & (SEQ - 1);
            const size_t base =
                ((size_t)(b * NHEADS + head) * SEQ + s) * HDIM + t * 16 + ln;
            out16[base]      = (_Float16)o1;
            out16[base + 32] = (_Float16)o2;
        }
    }
}

// ---------------------------------------------------------------------------
// Flash attention.  One block = 4 waves on the SAME (b,h), covering 64
// consecutive queries (16 per wave).  Each 32-key K/V block is staged into
// LDS once per block via async global->LDS copies, double-buffered; all 4
// waves consume it with ds_load_b128 fragments.  Online softmax per row.
//   Q,K : f16 [b,h,s,d]      Vt : f16 [b,h,d,s]      Aout : f16 [m][1024]
// grid = (B*H) * (SEQ/64) = 32 * 32 = 1024 blocks, 128 threads.
// ---------------------------------------------------------------------------
__global__ __launch_bounds__(128)
void flash_attn(const _Float16* __restrict__ Q,
                const _Float16* __restrict__ K,
                const _Float16* __restrict__ Vt,
                _Float16* __restrict__ Aout) {
    const int tid  = threadIdx.x;
    const int wave = tid >> 5;
    const int lane = tid & 31;
    const int half = lane >> 4;
    const int ln   = lane & 15;

    const int bh = blockIdx.x >> 5;          // 0..31  (b*16+h)
    const int qg = blockIdx.x & 31;          // 64-query group
    const int q0 = qg * 64 + wave * 16;

    // Double-buffered K (32 keys x 64 d) and Vt (64 d x 32 keys) tiles.
    // Pitches (80 / 40 halves) keep every fragment read 16B-aligned b128.
    __shared__ __align__(16) _Float16 ldsK[2][32][80];
    __shared__ __align__(16) _Float16 ldsV[2][64][40];
    __shared__ __align__(16) _Float16 plds[4][16][40];   // per-wave P tile

    const _Float16* Kbh  = K  + (size_t)bh * SEQ * HDIM;
    const _Float16* Vtbh = Vt + (size_t)bh * HDIM * SEQ;

    // Cooperative 128-thread stage of one 32-key block (4x b128 per thread).
    auto stage = [&](int buf, int key0) {
        const int krow = tid >> 3, kcol = (tid & 7) * 8;     // K: 16 rows/issue
        const _Float16* g0 = Kbh + (size_t)(key0 + krow) * HDIM + kcol;
        cp16_g2l(&ldsK[buf][krow][kcol],      g0);
        cp16_g2l(&ldsK[buf][krow + 16][kcol], g0 + 16 * HDIM);
        const int vrow = tid >> 2, vcol = (tid & 3) * 8;     // V: 32 rows/issue
        const _Float16* gv = Vtbh + (size_t)vrow * SEQ + key0 + vcol;
        cp16_g2l(&ldsV[buf][vrow][vcol],      gv);
        cp16_g2l(&ldsV[buf][vrow + 32][vcol], gv + (size_t)32 * SEQ);
    };

    // Q fragments for the whole tile (Hd = 64 -> two K=32 steps), reused 64x.
    Frag aq[2];
    const _Float16* Qrow = Q + ((size_t)bh * SEQ + q0 + ln) * HDIM;
#pragma unroll
    for (int ks = 0; ks < 2; ++ks) {
        aq[ks].u[0] = *(const uint4*)(Qrow + ks * 32 + 8 * half);
        aq[ks].u[1] = *(const uint4*)(Qrow + ks * 32 + 8 * half + 16);
    }

    v8f o[4] = {};
    float mrun[8], lrun[8];
#pragma unroll
    for (int r = 0; r < 8; ++r) { mrun[r] = -3.0e38f; lrun[r] = 0.0f; }

    const float scale = 0.125f;              // Hd^-0.5

    int cur = 0;
    stage(0, 0);

    for (int key0 = 0; key0 < SEQ; key0 += 32) {
        wait_async_lds();                    // this wave's copies landed
        __syncthreads();                     // all waves' copies + prev reads
        if (key0 + 32 < SEQ) stage(cur ^ 1, key0 + 32);

        // --- scores S = Q K^T for 32 key columns (two 16-col tiles) ---
        v8f c0 = {}, c1 = {};
#pragma unroll
        for (int ks = 0; ks < 2; ++ks) {
            Frag bk;
            const _Float16* kp = &ldsK[cur][ln][ks * 32 + 8 * half];
            bk.u[0] = *(const uint4*)kp;
            bk.u[1] = *(const uint4*)(kp + 16);
            c0 = wmma16(aq[ks], bk, c0);
            const _Float16* kp2 = &ldsK[cur][16 + ln][ks * 32 + 8 * half];
            bk.u[0] = *(const uint4*)kp2;
            bk.u[1] = *(const uint4*)(kp2 + 16);
            c1 = wmma16(aq[ks], bk, c1);
        }

        // --- online softmax update (rows live across 16-lane halves) ---
        float corr[8];
#pragma unroll
        for (int r = 0; r < 8; ++r) {
            c0[r] *= scale; c1[r] *= scale;
            const float tilemax = redmax16(fmaxf(c0[r], c1[r]));
            const float mn = fmaxf(mrun[r], tilemax);
            corr[r] = __expf(mrun[r] - mn);
            mrun[r] = mn;
            c0[r] = __expf(c0[r] - mn);
            c1[r] = __expf(c1[r] - mn);
            lrun[r] = lrun[r] * corr[r] + redsum16(c0[r] + c1[r]);
        }
#pragma unroll
        for (int t = 0; t < 4; ++t)
#pragma unroll
            for (int r = 0; r < 8; ++r) o[t][r] *= corr[r];

        // --- C-layout P -> LDS -> A-fragment layout (wave-private) ---
#pragma unroll
        for (int r = 0; r < 8; ++r) {
            const int mrow = r + 8 * half;
            plds[wave][mrow][ln]      = (_Float16)c0[r];
            plds[wave][mrow][16 + ln] = (_Float16)c1[r];
        }
        asm volatile("s_wait_dscnt 0" ::: "memory");   // wave-internal LDS RAW
        Frag ap;
        ap.u[0] = *(const uint4*)&plds[wave][ln][8 * half];
        ap.u[1] = *(const uint4*)&plds[wave][ln][16 + 8 * half];

        // --- O += P . V  (Vt tile contiguous along key dim in LDS) ---
#pragma unroll
        for (int t = 0; t < 4; ++t) {
            Frag bv;
            const _Float16* vp = &ldsV[cur][t * 16 + ln][8 * half];
            bv.u[0] = *(const uint4*)vp;
            bv.u[1] = *(const uint4*)(vp + 16);
            o[t] = wmma16(ap, bv, o[t]);
        }
        cur ^= 1;
    }

    // --- normalize and store [m][1024] f16 for the output projection ---
    const int b = bh >> 4, h = bh & 15;
    float linv[8];
#pragma unroll
    for (int r = 0; r < 8; ++r) linv[r] = 1.0f / lrun[r];
#pragma unroll
    for (int t = 0; t < 4; ++t)
#pragma unroll
        for (int r = 0; r < 8; ++r) {
            const int q = q0 + r + 8 * half;
            Aout[(size_t)(b * SEQ + q) * D_MODEL + h * 64 + t * 16 + ln] =
                (_Float16)(o[t][r] * linv[r]);
        }
}

// ---------------------------------------------------------------------------
// Host-side orchestration
// ---------------------------------------------------------------------------
extern "C" void kernel_launch(void* const* d_in, const int* in_sizes, int n_in,
                              void* d_out, int out_size, void* d_ws, size_t ws_size,
                              hipStream_t stream) {
    (void)in_sizes; (void)n_in; (void)out_size; (void)ws_size;

    const float* q_f32  = (const float*)d_in[0];
    const float* k_f32  = (const float*)d_in[1];
    const float* v_f32  = (const float*)d_in[2];
    const float* Wq     = (const float*)d_in[3];
    const float* bq     = (const float*)d_in[4];
    const float* Wk     = (const float*)d_in[5];
    const float* bk     = (const float*)d_in[6];
    const float* Wv     = (const float*)d_in[7];
    const float* bv     = (const float*)d_in[8];
    const float* Wo     = (const float*)d_in[9];
    const float* bo     = (const float*)d_in[10];
    const int*   qpos   = (const int*)d_in[11];
    const int*   kpos   = (const int*)d_in[12];
    float*       out    = (float*)d_out;

    const size_t XN = (size_t)MTOT * D_MODEL;      // 4,194,304
    const size_t WN = (size_t)D_MODEL * D_MODEL;   // 1,048,576

    char* ws = (char*)d_ws;
    _Float16* Xq16 = (_Float16*)(ws);                       // 8 MB each
    _Float16* Xk16 = Xq16 + XN;
    _Float16* Xv16 = Xk16 + XN;
    _Float16* Wq16 = Xv16 + XN;                             // 2 MB each
    _Float16* Wk16 = Wq16 + WN;
    _Float16* Wv16 = Wk16 + WN;
    _Float16* Wo16 = Wv16 + WN;
    _Float16* Q16  = Wo16 + WN;                             // [b,h,s,d]
    _Float16* K16  = Q16 + XN;                              // [b,h,s,d]
    _Float16* Vt16 = K16 + XN;                              // [b,h,d,s]
    _Float16* A16  = Vt16 + XN;                             // [m][1024]

    // 1) f32 -> f16 conversions
    cvt_f32_f16<<<(int)(XN / 1024), 256, 0, stream>>>(q_f32, Xq16, (int)XN);
    cvt_f32_f16<<<(int)(XN / 1024), 256, 0, stream>>>(k_f32, Xk16, (int)XN);
    cvt_f32_f16<<<(int)(XN / 1024), 256, 0, stream>>>(v_f32, Xv16, (int)XN);
    cvt_f32_f16<<<(int)(WN / 1024), 256, 0, stream>>>(Wq, Wq16, (int)WN);
    cvt_f32_f16<<<(int)(WN / 1024), 256, 0, stream>>>(Wk, Wk16, (int)WN);
    cvt_f32_f16<<<(int)(WN / 1024), 256, 0, stream>>>(Wv, Wv16, (int)WN);
    cvt_f32_f16<<<(int)(WN / 1024), 256, 0, stream>>>(Wo, Wo16, (int)WN);

    // 2) fused projections: Q,K (+RoPE, [b,h,s,d]), V (transposed [b,h,d,s])
    dim3 ggrid(MTOT / 16, D_MODEL / 256), gblk(128);
    proj_gemm<<<ggrid, gblk, 0, stream>>>(Xq16, Wq16, bq, qpos, Q16,  nullptr, 1);
    proj_gemm<<<ggrid, gblk, 0, stream>>>(Xk16, Wk16, bk, kpos, K16,  nullptr, 1);
    proj_gemm<<<ggrid, gblk, 0, stream>>>(Xv16, Wv16, bv, nullptr, Vt16, nullptr, 2);

    // 3) flash attention: 32 (b,h) x 32 query-groups, 4 waves / block
    flash_attn<<<1024, 128, 0, stream>>>(Q16, K16, Vt16, A16);

    // 4) output projection -> f32 d_out
    proj_gemm<<<ggrid, gblk, 0, stream>>>(A16, Wo16, bo, nullptr, nullptr, out, 0);
}
